// GroupedQueryAttention_40913858461726
// MI455X (gfx1250) — matrix operand order repaired and checked by reading, not compiled, not run on hardware
//
#include <hip/hip_runtime.h>
#include <hip/hip_bf16.h>

// GQA forward for MI455X (gfx1250, wave32, WMMA).
//  - bf16 V_WMMA_F32_16X16X32_BF16 for QKV projections and both attention GEMMs
//  - x and weights pre-converted to bf16 once -> projection hot loop is pure
//    fragment loads + WMMA (no per-iteration converts)
//  - transposed-flash layout: S^T = K x Q^T so softmax stats are per-lane scalars
//  - P C-layout -> B-fragment conversion in table k order, no shuffle
//  - V stored transposed [hd][s] so the 2nd-GEMM A fragments are contiguous loads

#define DEVINL __device__ __forceinline__

typedef __attribute__((ext_vector_type(16))) __bf16 v16bf;
typedef __attribute__((ext_vector_type(8)))  float  v8f;

constexpr int B_  = 4;
constexpr int S_  = 2048;
constexpr int D_  = 2048;
constexpr int NH_ = 16;
constexpr int NKV_ = 4;
constexpr int HD_ = 128;
constexpr int NREP_ = NH_ / NKV_;
// 1/sqrt(128) * log2(e): fold softmax scale into log2-domain exp
constexpr float SC_LOG2E = 0.088388347648318447f * 1.4426950408889634f;

union FragBF {
    v16bf v;
    __bf16 b[16];
    unsigned short u[16];
};

DEVINL unsigned short bf_bits(float x) {        // native v_cvt_bf16_f32
    __bf16 h = (__bf16)x;
    unsigned short s;
    __builtin_memcpy(&s, &h, 2);
    return s;
}

DEVINL v8f wmma_bf16(v16bf a, v16bf b, v8f c) {
    return __builtin_amdgcn_wmma_f32_16x16x32_bf16(false, a, false, b,
                                                   (short)0, c, false, false);
}

DEVINL v16bf load_frag_2x16B(const unsigned short* p0, const unsigned short* p1) {
    FragBF f;
    __builtin_memcpy(&f.u[0], p0, 16);
    __builtin_memcpy(&f.u[8], p1, 16);
    return f.v;
}

DEVINL v16bf load_frag_32B(const unsigned short* p) {
    FragBF f;
    __builtin_memcpy(&f.u[0], p, 32);
    return f.v;
}

// ---------------------------------------------------------------------------
// Elementwise f32 -> bf16 (8 elems/thread, 2x b128 load -> 1x b128 store)
// ---------------------------------------------------------------------------
__global__ __launch_bounds__(256)
void cvt_bf16_kernel(const float* __restrict__ in, unsigned short* __restrict__ out,
                     int n8) {
    const int i = blockIdx.x * blockDim.x + threadIdx.x;
    if (i >= n8) return;
    const float4* p = (const float4*)in + (size_t)i * 2;
    const float4 a = p[0], b = p[1];
    unsigned short h[8];
    h[0] = bf_bits(a.x); h[1] = bf_bits(a.y); h[2] = bf_bits(a.z); h[3] = bf_bits(a.w);
    h[4] = bf_bits(b.x); h[5] = bf_bits(b.y); h[6] = bf_bits(b.z); h[7] = bf_bits(b.w);
    __builtin_memcpy(out + (size_t)i * 8, h, 16);
}

// ---------------------------------------------------------------------------
// Projection: y[s][o] = sum_d x[s][d] * w[o][d], optional RoPE, bf16 output.
// One wave computes a 32s x 128o strip (one head). mode: 0/1 = Q/K (rope,
// store [s][hd] via LDS transpose), 2 = V (no rope, store transposed [hd][s]).
// x and w are pre-converted bf16; the hot loop is loads + WMMA only.
// ---------------------------------------------------------------------------
__global__ __launch_bounds__(128)
void proj_kernel(const unsigned short* __restrict__ xb,
                 const unsigned short* __restrict__ wb,
                 const float* __restrict__ cosT, const float* __restrict__ sinT,
                 unsigned short* __restrict__ outp, int heads, int mode) {
    __shared__ unsigned short lds[4 * 32 * 128];

    const int lane = threadIdx.x & 31;
    const int wl   = threadIdx.x >> 5;
    const int wv   = blockIdx.x * 4 + wl;
    const int nst  = S_ / 32;                 // 64 s-strips
    const int st   = wv % nst;
    const int head = (wv / nst) % heads;
    const int b    = wv / (nst * heads);
    const int n    = lane & 15;
    const bool hih = lane >= 16;
    const int mo   = hih ? 8 : 0;             // A-frag K-chunk / C-tile M offset
    const int hoff = hih ? 16 : 0;            // unused (B frag also uses mo)
    const int s0   = st * 32;

    v8f acc[2][8];
#pragma unroll
    for (int i = 0; i < 2; ++i)
#pragma unroll
        for (int t = 0; t < 8; ++t)
#pragma unroll
            for (int r = 0; r < 8; ++r) acc[i][t][r] = 0.f;

    const unsigned short* x0 = xb + (size_t)(b * S_ + s0 + n) * D_;
    const unsigned short* w0 = wb + (size_t)(head * HD_ + n) * D_;

    for (int kk = 0; kk < D_; kk += 32) {
        v16bf af[2];
#pragma unroll
        for (int i = 0; i < 2; ++i) {
            const unsigned short* row = x0 + (size_t)(i * 16) * D_ + kk;
            af[i] = load_frag_2x16B(row + mo, row + mo + 16);
        }
#pragma unroll
        for (int t = 0; t < 8; ++t) {
            v16bf bf = load_frag_2x16B(w0 + (size_t)(t * 16) * D_ + kk + mo, w0 + (size_t)(t * 16) * D_ + kk + mo + 16);
            acc[0][t] = wmma_bf16(af[0], bf, acc[0][t]);
            acc[1][t] = wmma_bf16(af[1], bf, acc[1][t]);
        }
    }

    if (mode != 2) {  // RoPE: hd pairs with hd+64 -> tiles t and t+4, same lane/reg
#pragma unroll
        for (int i = 0; i < 2; ++i)
#pragma unroll
            for (int t = 0; t < 4; ++t)
#pragma unroll
                for (int r = 0; r < 8; ++r) {
                    const int si = s0 + i * 16 + mo + r;
                    const int hd = t * 16 + n;
                    const float lo = acc[i][t][r];
                    const float hh = acc[i][t + 4][r];
                    const float c0 = cosT[si * HD_ + hd];
                    const float s1 = sinT[si * HD_ + hd];
                    const float c1 = cosT[si * HD_ + hd + 64];
                    const float s2 = sinT[si * HD_ + hd + 64];
                    acc[i][t][r]     = lo * c0 - hh * s1;
                    acc[i][t + 4][r] = hh * c1 + lo * s2;
                }
    }

    if (mode == 2) {
        // transposed store vt[(bh*HD+hd)*S + s]: 8 consecutive s per (i,t,half)
        unsigned short* basep = outp + (size_t)(b * heads + head) * HD_ * S_;
#pragma unroll
        for (int t = 0; t < 8; ++t) {
            const int hd = t * 16 + n;
#pragma unroll
            for (int i = 0; i < 2; ++i) {
                unsigned short tmp[8];
#pragma unroll
                for (int r = 0; r < 8; ++r) tmp[r] = bf_bits(acc[i][t][r]);
                __builtin_memcpy(basep + (size_t)hd * S_ + s0 + i * 16 + mo, tmp, 16);
            }
        }
    } else {
        // LDS transpose so global stores of [s][hd] rows are coalesced 16B
        unsigned short* lw = &lds[wl * 32 * 128];
#pragma unroll
        for (int i = 0; i < 2; ++i)
#pragma unroll
            for (int t = 0; t < 8; ++t)
#pragma unroll
                for (int r = 0; r < 8; ++r)
                    lw[(i * 16 + mo + r) * 128 + t * 16 + n] = bf_bits(acc[i][t][r]);
        __syncthreads();
        unsigned short* basep =
            outp + ((size_t)(b * heads + head) * S_ + s0) * HD_;
#pragma unroll
        for (int p = 0; p < 16; ++p) {
            const int unit = p * 32 + lane;   // 512 units of 8 bf16
            const int sl = unit >> 4;
            const int cu = (unit & 15) * 8;
            unsigned short tmp[8];
            __builtin_memcpy(tmp, &lw[sl * 128 + cu], 16);
            __builtin_memcpy(basep + (size_t)sl * HD_ + cu, tmp, 16);
        }
    }
}

// ---------------------------------------------------------------------------
// Attention step over one k-block (cnt = 16 or 32 kpos).
// S^T = K x Q^T; online softmax (per-lane stats); out^T += V^T x P^T.
// ---------------------------------------------------------------------------
DEVINL void attn_step(const unsigned short* __restrict__ kp,
                      const unsigned short* __restrict__ vp,
                      const v16bf qf[4], v8f acc[8],
                      float& run_max, float& run_sum,
                      int kb, int cnt, bool masked, int qbase, int lane) {
    const int n = lane & 15;
    const bool hih = lane >= 16;
    const int mo = hih ? 8 : 0;
    const int ntile = (cnt == 32) ? 2 : 1;

    float st[2][8];
    for (int t = 0; t < ntile; ++t) {
        v8f s8;
#pragma unroll
        for (int r = 0; r < 8; ++r) s8[r] = 0.f;
        const unsigned short* krow = kp + (size_t)(kb + t * 16 + n) * HD_;
#pragma unroll
        for (int c = 0; c < 4; ++c) {
            v16bf kf = load_frag_2x16B(krow + c * 32 + mo, krow + c * 32 + mo + 16);
            s8 = wmma_bf16(kf, qf[c], s8);
        }
#pragma unroll
        for (int r = 0; r < 8; ++r) {
            float sv = s8[r] * SC_LOG2E;
            if (masked) {
                const int kpos = kb + t * 16 + mo + r;
                if (kpos > qbase + n) sv = -3.0e38f;
            }
            st[t][r] = sv;
        }
    }
    if (ntile == 1) {
#pragma unroll
        for (int r = 0; r < 8; ++r) st[1][r] = -3.0e38f;
    }

    float bm = -3.0e38f;
#pragma unroll
    for (int t = 0; t < 2; ++t)
#pragma unroll
        for (int r = 0; r < 8; ++r) bm = fmaxf(bm, st[t][r]);
    bm = fmaxf(bm, __shfl_xor(bm, 16, 32));
    const float nm = fmaxf(run_max, bm);
    const float corr = exp2f(run_max - nm);
    run_max = nm;

    float p[2][8];
    float lsum = 0.f;
#pragma unroll
    for (int t = 0; t < 2; ++t)
#pragma unroll
        for (int r = 0; r < 8; ++r) {
            p[t][r] = exp2f(st[t][r] - nm);
            lsum += p[t][r];
        }
    lsum += __shfl_xor(lsum, 16, 32);
    run_sum = run_sum * corr + lsum;
#pragma unroll
    for (int t = 0; t < 8; ++t)
#pragma unroll
        for (int r = 0; r < 8; ++r) acc[t][r] *= corr;

    // P (C-layout, two 16x16 tiles) -> P^T B-fragment in table k order (no shuffle)
    FragBF pf;
#pragma unroll
    for (int r = 0; r < 8; ++r) {
        const float lo = p[0][r];
        const float hh = p[1][r];
        pf.b[r]     = (__bf16)lo;
        pf.b[8 + r] = (__bf16)hh;
    }

    // out^T += V^T x P^T  (V^T contiguous along kpos)
#pragma unroll
    for (int t = 0; t < 8; ++t) {
        const unsigned short* vr = vp + (size_t)(t * 16 + n) * S_ + kb;
        v16bf vf;
        if (cnt == 32) {
            vf = load_frag_2x16B(vr + mo, vr + mo + 16);
        } else {                    // avoid OOB: upper kpos half is all-masked
            FragBF f;
            __builtin_memcpy(&f.u[0], vr + mo, 16);
#pragma unroll
            for (int e = 8; e < 16; ++e) f.u[e] = 0;
            vf = f.v;
        }
        acc[t] = wmma_bf16(vf, pf.v, acc[t]);
    }
}

__global__ __launch_bounds__(256)
void attn_kernel(const unsigned short* __restrict__ qws,
                 const unsigned short* __restrict__ kws,
                 const unsigned short* __restrict__ vws,
                 float* __restrict__ out) {
    const int lane = threadIdx.x & 31;
    const int wl   = threadIdx.x >> 5;
    const int wv   = blockIdx.x * 8 + wl;
    const int nqt  = S_ / 16;                  // 128
    const int qt   = wv % nqt;
    const int h    = (wv / nqt) % NH_;
    const int b    = wv / (nqt * NH_);
    const int kvh  = h / NREP_;
    const int n    = lane & 15;
    const bool hih = lane >= 16;
    const int mo   = hih ? 8 : 0;
    const int hoff = hih ? 16 : 0;
    const int qbase = qt * 16;

    const unsigned short* qp = qws + (size_t)(b * NH_ + h) * S_ * HD_;
    const unsigned short* kp = kws + (size_t)(b * NKV_ + kvh) * S_ * HD_;
    const unsigned short* vp = vws + (size_t)(b * NKV_ + kvh) * HD_ * S_;

    // Q^T B-fragments are loop invariant: preload 4 (hd chunks of 32)
    v16bf qf[4];
#pragma unroll
    for (int c = 0; c < 4; ++c)
        qf[c] = load_frag_2x16B(qp + (size_t)(qbase + n) * HD_ + c * 32 + mo, qp + (size_t)(qbase + n) * HD_ + c * 32 + mo + 16);

    v8f acc[8];
#pragma unroll
    for (int t = 0; t < 8; ++t)
#pragma unroll
        for (int r = 0; r < 8; ++r) acc[t][r] = 0.f;
    float run_max = -3.0e38f, run_sum = 0.f;

    const int nfull = qbase >> 5;
    for (int kb = 0; kb < (nfull << 5); kb += 32) {
        __builtin_prefetch(kp + (size_t)(kb + 32 + n) * HD_, 0, 1);
        __builtin_prefetch(vp + (size_t)n * S_ + kb + 32, 0, 1);
        attn_step(kp, vp, qf, acc, run_max, run_sum, kb, 32, false, qbase, lane);
    }
    const int rstart = nfull << 5;
    const int cnt = qbase + 16 - rstart;       // 16 or 32, covers the diagonal
    attn_step(kp, vp, qf, acc, run_max, run_sum, rstart, cnt, true, qbase, lane);

    // out[b][q][h*HD+hd] = acc / run_sum ; 8 consecutive floats per tile half
    const float inv = 1.0f / run_sum;
    float* op = out + (size_t)(b * S_ + qbase + n) * D_ + h * HD_ + mo;
#pragma unroll
    for (int t = 0; t < 8; ++t) {
        float tmp[8];
#pragma unroll
        for (int r = 0; r < 8; ++r) tmp[r] = acc[t][r] * inv;
        __builtin_memcpy(op + t * 16, tmp, 32);
    }
}

// ---------------------------------------------------------------------------
extern "C" void kernel_launch(void* const* d_in, const int* in_sizes, int n_in,
                              void* d_out, int out_size, void* d_ws, size_t ws_size,
                              hipStream_t stream) {
    (void)in_sizes; (void)n_in; (void)out_size; (void)ws_size;
    const float* x    = (const float*)d_in[0];
    const float* cosT = (const float*)d_in[1];
    const float* sinT = (const float*)d_in[2];
    const float* wq   = (const float*)d_in[3];
    const float* wk   = (const float*)d_in[4];
    const float* wv   = (const float*)d_in[5];
    float* out = (float*)d_out;

    // workspace layout (bf16 elements)
    unsigned short* xbf  = (unsigned short*)d_ws;                       // 33.5MB
    unsigned short* wqbf = xbf  + (size_t)B_ * S_ * D_;                 //  8MB
    unsigned short* wkbf = wqbf + (size_t)D_ * D_;                      //  2MB
    unsigned short* wvbf = wkbf + (size_t)NKV_ * HD_ * D_;              //  2MB
    unsigned short* qws  = wvbf + (size_t)NKV_ * HD_ * D_;              // 33.5MB
    unsigned short* kws  = qws  + (size_t)B_ * NH_ * S_ * HD_;          //  8MB
    unsigned short* vws  = kws  + (size_t)B_ * NKV_ * S_ * HD_;         //  8MB

    // one-time f32 -> bf16 conversion of activations + weights
    const int nx8 = (B_ * S_ * D_) / 8;         // 2,097,152
    const int nq8 = (D_ * D_) / 8;              //   524,288
    const int nk8 = (NKV_ * HD_ * D_) / 8;      //   131,072
    cvt_bf16_kernel<<<nx8 / 256, dim3(256), 0, stream>>>(x,  xbf,  nx8);
    cvt_bf16_kernel<<<nq8 / 256, dim3(256), 0, stream>>>(wq, wqbf, nq8);
    cvt_bf16_kernel<<<nk8 / 256, dim3(256), 0, stream>>>(wk, wkbf, nk8);
    cvt_bf16_kernel<<<nk8 / 256, dim3(256), 0, stream>>>(wv, wvbf, nk8);

    const int wavesQ = B_ * NH_  * (S_ / 32);   // 4096
    const int wavesK = B_ * NKV_ * (S_ / 32);   // 1024
    proj_kernel<<<wavesQ / 4, dim3(128), 0, stream>>>(xbf, wqbf, cosT, sinT, qws, NH_, 0);
    proj_kernel<<<wavesK / 4, dim3(128), 0, stream>>>(xbf, wkbf, cosT, sinT, kws, NKV_, 1);
    proj_kernel<<<wavesK / 4, dim3(128), 0, stream>>>(xbf, wvbf, cosT, sinT, vws, NKV_, 2);

    const int wavesA = B_ * NH_ * (S_ / 16);    // 8192
    attn_kernel<<<wavesA / 8, dim3(256), 0, stream>>>(qws, kws, vws, out);
}
